// SeqGraph_27986006901054
// MI455X (gfx1250) — matrix operand order repaired and checked by reading, not compiled, not run on hardware
//
#include <hip/hip_runtime.h>
#include <hip/hip_bf16.h>
#include <cstdint>

typedef __attribute__((ext_vector_type(16))) __bf16       v16bf;
typedef __attribute__((ext_vector_type(8)))  float        v8f;
typedef __attribute__((ext_vector_type(8)))  unsigned int v8u;

#define WMMA_BF16(a, b, c) \
  __builtin_amdgcn_wmma_f32_16x16x32_bf16(false, (a), false, (b), (short)0, (c), false, false)

// ---- fragment layouts used here (contiguous k = half*16 + s, not the ISA table order; wave32) ----
// Every fragment VGPR holds two bf16 that are K-adjacent in memory, so each
// VGPR is one aligned dword load (groups of 4 merge into b128).
//
// A 16x32 bf16: lane l (m = l&15, half = l>>4); vgpr v holds
//   K = half*16 + v*2  .. +1
__device__ __forceinline__ v16bf sg_load_a(const __bf16* row, int kbase, int half) {
  v8u u;
#pragma unroll
  for (int v = 0; v < 8; ++v) {
    int k = kbase + (half << 4) + (v << 1);
    u[v] = *(const unsigned int*)(row + k);
  }
  return __builtin_bit_cast(v16bf, u);
}
// B 32x16 bf16: lane l (n = l&15, half = l>>4); vgpr v holds
//   K = half*16 + v*2 .. +1, for column n.
// X tile row-major [n][128]: B[k][n] = X[n][k] -> 8 contiguous dwords per frag.
__device__ __forceinline__ v16bf sg_load_b_xrow(const __bf16* xt, int n,
                                                int kbase, int half) {
  v8u u;
#pragma unroll
  for (int v = 0; v < 8; ++v)
    u[v] = *(const unsigned int*)(xt + n * 128 + kbase + (half << 4) + (v << 1));
  return __builtin_bit_cast(v16bf, u);
}
// Weights pre-packed pair-interleaved: Wp[k/2][col][2] (128 cols) -> one dword
// per VGPR per lane; adjacent lanes (cols) stay coalesced.
__device__ __forceinline__ v16bf sg_load_b_kp(const __bf16* matp, int col,
                                              int kbase, int half) {
  v8u u;
#pragma unroll
  for (int v = 0; v < 8; ++v) {
    int kp = (kbase >> 1) + (half << 3) + v;
    u[v] = *(const unsigned int*)(matp + kp * 256 + col * 2);
  }
  return __builtin_bit_cast(v16bf, u);
}

// ---- prep: symmetric leaky adjacency, z_i = A^i z chain, packed bf16 weights ----
__global__ void sg_prep(const float* __restrict__ hidden_adj,
                        const float* __restrict__ hidden_feat,
                        const float* __restrict__ fc_w,
                        const float* __restrict__ mlp_w,
                        float* __restrict__ Zf, __bf16* __restrict__ Zh,
                        __bf16* __restrict__ Whp, __bf16* __restrict__ Mhp) {
  __shared__ float As[16 * 100];
  int tid = threadIdx.x;
  for (int i = tid; i < 1600; i += 256) As[i] = 0.f;
  __syncthreads();
  for (int i = tid; i < 16 * 45; i += 256) {
    int g = i / 45, p = i % 45;
    int r = 0, rem = p;
    while (rem >= 9 - r) { rem -= 9 - r; ++r; }
    int cidx = r + 1 + rem;
    float v = hidden_adj[i];
    v = v >= 0.f ? v : 0.01f * v;               // LeakyReLU
    As[g * 100 + r * 10 + cidx] = v;
    As[g * 100 + cidx * 10 + r] = v;            // symmetrize
  }
  for (int i = tid; i < 20480; i += 256) {      // z0
    float v = hidden_feat[i];
    Zf[i] = v; Zh[i] = (__bf16)v;
  }
  for (int i = tid; i < 128 * 128; i += 256) {  // fc_w -> pair-interleaved bf16
    int k = i >> 7, col = i & 127;
    Whp[(k >> 1) * 256 + col * 2 + (k & 1)] = (__bf16)fc_w[i];
  }
  for (int i = tid; i < 64 * 128; i += 256) {   // mlp_w -> pair-interleaved bf16
    int k = i >> 7, col = i & 127;
    Mhp[(k >> 1) * 256 + col * 2 + (k & 1)] = (__bf16)mlp_w[i];
  }
  __syncthreads();
  for (int s = 1; s < 4; ++s) {                 // z_s = A z_{s-1}
    const float* zp = Zf + (s - 1) * 20480;
    for (int i = tid; i < 20480; i += 256) {
      int gs = i >> 7, d = i & 127;
      int g = gs / 10, srow = gs % 10;
      float acc = 0.f;
#pragma unroll
      for (int t = 0; t < 10; ++t)
        acc += As[g * 100 + srow * 10 + t] * zp[(g * 10 + t) * 128 + d];
      Zf[s * 20480 + i] = acc;
      Zh[s * 20480 + i] = (__bf16)acc;
    }
    __syncthreads();
  }
}

// ---- x0 = sigmoid(gather(emb) @ W + b): WMMA, 16 nodes x 128 cols / block ----
__global__ void sg_embed_fc(const int* __restrict__ x_idx,
                            const float* __restrict__ poi_emb,
                            const __bf16* __restrict__ Whp,
                            const float* __restrict__ fc_b,
                            float* __restrict__ x0f, __bf16* __restrict__ x0h) {
  __shared__ __bf16 lA[16 * 128];
  int tid = threadIdx.x;
  int base = blockIdx.x * 16;
  for (int i = tid; i < 2048; i += 256) {
    int n = i >> 7, d = i & 127;
    int row = x_idx[base + n];
    lA[i] = (__bf16)poi_emb[(size_t)row * 128 + d];
  }
  __syncthreads();
  int wave = tid >> 5, lane = tid & 31;
  int half = lane >> 4, lm = lane & 15;
  int col = wave * 16 + lm;
  v8f c = {};
#pragma unroll
  for (int kc = 0; kc < 4; ++kc) {
    int kb = kc * 32;
    v16bf a = sg_load_a(lA + lm * 128, kb, half);
    v16bf b = sg_load_b_kp(Whp, col, kb, half);
    c = WMMA_BF16(a, b, c);
  }
  float bias = fc_b[col];
#pragma unroll
  for (int v = 0; v < 8; ++v) {
    int m = v + (half << 3);
    size_t node = (size_t)(base + m);
    float t = c[v] + bias;
    float s = __builtin_amdgcn_rcpf(1.f + __expf(-t));  // fast sigmoid (v_rcp_f32)
    x0f[node * 128 + col] = s;
    x0h[node * 128 + col] = (__bf16)s;
  }
}

// ---- fused zx0*zx_i -> segment-pool into outacc[B][64]; 10 waves = 10 gs-tiles ----
// X tiles staged via async Global->LDS DMA (ASYNCcnt path, no VGPR round-trip).
__global__ void sg_zx_fuse(const __bf16* __restrict__ Zh,
                           const __bf16* __restrict__ x0h,
                           const __bf16* __restrict__ xih,
                           const int* __restrict__ batch,
                           float* __restrict__ outacc, int step) {
  __shared__ __bf16 lx0[16 * 128];
  __shared__ __bf16 lxi[16 * 128];
  int tid = threadIdx.x;
  int base = blockIdx.x * 16;
  // 256 lanes x 16B x 2 tiles = 8KB staged; LDS byte offset = low 32 bits of
  // generic shared pointer (aperture rule: addr[31:0] is the LDS offset).
  if (tid < 256) {
    int n = tid >> 4, ch = tid & 15;
    const __bf16* g0 = x0h + (size_t)(base + n) * 128 + ch * 8;
    const __bf16* g1 = xih + (size_t)(base + n) * 128 + ch * 8;
    unsigned l0 = (unsigned)(uintptr_t)(lx0 + n * 128 + ch * 8);
    unsigned l1 = (unsigned)(uintptr_t)(lxi + n * 128 + ch * 8);
    asm volatile("global_load_async_to_lds_b128 %0, %1, off\n\t"
                 "global_load_async_to_lds_b128 %2, %3, off"
                 :: "v"(l0), "v"(g0), "v"(l1), "v"(g1)
                 : "memory");
    asm volatile("s_wait_asynccnt 0" ::: "memory");
  }
  __syncthreads();
  int gt = tid >> 5, lane = tid & 31;
  int half = lane >> 4, lm = lane & 15;
  const __bf16* Z0 = Zh;
  const __bf16* Zi = Zh + step * 20480;
  v8f c0 = {}, ci = {};
#pragma unroll
  for (int kc = 0; kc < 4; ++kc) {
    int kb = kc * 32;
    v16bf a0 = sg_load_a(Z0 + (gt * 16 + lm) * 128, kb, half);
    v16bf b0 = sg_load_b_xrow(lx0, lm, kb, half);
    c0 = WMMA_BF16(a0, b0, c0);
    v16bf ai = sg_load_a(Zi + (gt * 16 + lm) * 128, kb, half);
    v16bf bi = sg_load_b_xrow(lxi, lm, kb, half);
    ci = WMMA_BF16(ai, bi, ci);
  }
  // lane column = node lm; rows m = half*8+v are 8 consecutive gs rows -> <=2 g's
  int b = batch[base + lm];
  float* obase = outacc + (size_t)b * 64 + step * 16;
  float acc = 0.f;
  int prevg = (gt * 16 + (half << 3)) / 10;
#pragma unroll
  for (int v = 0; v < 8; ++v) {
    int g = (gt * 16 + (half << 3) + v) / 10;
    if (g != prevg) { atomicAdd(obase + prevg, acc); acc = 0.f; prevg = g; }
    acc += c0[v] * ci[v];
  }
  atomicAdd(obase + prevg, acc);
}

// ---- edge scatter-add: one wave per edge, 4 dims per lane ----
__global__ void sg_mp(const int* __restrict__ edge_index,
                      const float* __restrict__ xcur,
                      float* __restrict__ xnext, int E_) {
  int tid = threadIdx.x;
  size_t e = (size_t)blockIdx.x * 8 + (tid >> 5);
  int lane = tid & 31;
  __builtin_prefetch(edge_index + e + 2048, 0, 0);        // global_prefetch_b8
  __builtin_prefetch(edge_index + E_ + e + 2048, 0, 0);
  int src = edge_index[e];
  int dst = edge_index[(size_t)E_ + e];
  const float4 val = *(const float4*)(xcur + (size_t)src * 128 + lane * 4);
  float* o = xnext + (size_t)dst * 128 + lane * 4;
  atomicAdd(o + 0, val.x);
  atomicAdd(o + 1, val.y);
  atomicAdd(o + 2, val.z);
  atomicAdd(o + 3, val.w);
}

__global__ void sg_cvt(const float* __restrict__ src, __bf16* __restrict__ dst,
                       size_t n4) {
  size_t i = ((size_t)blockIdx.x * blockDim.x + threadIdx.x);
  if (i < n4) {
    float4 v = *(const float4*)(src + i * 4);
    size_t o = i * 4;
    dst[o + 0] = (__bf16)v.x;
    dst[o + 1] = (__bf16)v.y;
    dst[o + 2] = (__bf16)v.z;
    dst[o + 3] = (__bf16)v.w;
  }
}

__global__ void sg_zero4(float4* __restrict__ p, size_t n4) {
  size_t i = (size_t)blockIdx.x * blockDim.x + threadIdx.x;
  if (i < n4) p[i] = make_float4(0.f, 0.f, 0.f, 0.f);
}

__global__ void sg_zero(float* __restrict__ p, size_t n) {
  size_t i = (size_t)blockIdx.x * blockDim.x + threadIdx.x;
  if (i < n) p[i] = 0.f;
}

// ---- final: leaky(outacc[B,64] @ mlp_w[64,128] + b) via WMMA ----
__global__ void sg_final(const float* __restrict__ outacc,
                         const __bf16* __restrict__ Mhp,
                         const float* __restrict__ mlp_b,
                         float* __restrict__ out) {
  __shared__ __bf16 lA[16 * 64];
  int tid = threadIdx.x;
  int rbase = blockIdx.x * 16;
  for (int i = tid; i < 1024; i += 256)
    lA[i] = (__bf16)outacc[(size_t)(rbase + (i >> 6)) * 64 + (i & 63)];
  __syncthreads();
  int wave = tid >> 5, lane = tid & 31;
  int half = lane >> 4, lm = lane & 15;
  int col = wave * 16 + lm;
  v8f c = {};
#pragma unroll
  for (int kc = 0; kc < 2; ++kc) {
    int kb = kc * 32;
    v16bf a = sg_load_a(lA + lm * 64, kb, half);
    v16bf b = sg_load_b_kp(Mhp, col, kb, half);
    c = WMMA_BF16(a, b, c);
  }
  float bias = mlp_b[col];
#pragma unroll
  for (int v = 0; v < 8; ++v) {
    int m = v + (half << 3);
    float t = c[v] + bias;
    out[(size_t)(rbase + m) * 128 + col] = t >= 0.f ? t : 0.01f * t;
  }
}

extern "C" void kernel_launch(void* const* d_in, const int* in_sizes, int n_in,
                              void* d_out, int out_size, void* d_ws, size_t ws_size,
                              hipStream_t stream) {
  const int*   x_idx      = (const int*)d_in[0];
  const int*   edge_index = (const int*)d_in[1];
  const int*   batch      = (const int*)d_in[2];
  const float* poi_emb    = (const float*)d_in[3];
  const float* fc_w       = (const float*)d_in[4];
  const float* fc_b       = (const float*)d_in[5];
  const float* hidden_adj = (const float*)d_in[6];
  const float* hidden_feat= (const float*)d_in[7];
  const float* mlp_w      = (const float*)d_in[8];
  const float* mlp_b      = (const float*)d_in[9];
  float* out = (float*)d_out;

  const int N_ = in_sizes[0];          // 100000 (multiple of 16)
  const int E_ = in_sizes[1] / 2;      // 640000 (multiple of 8)
  const int B_ = 2048;
  const size_t NB = (size_t)N_ * 128;

  char* ws = (char*)d_ws;
  size_t off = 0;
  auto wsa = [&](size_t bytes) -> char* {
    char* p = ws + off;
    off += (bytes + 255) & ~(size_t)255;
    return p;
  };
  float*  xa     = (float*)wsa(NB * 4);
  float*  xb     = (float*)wsa(NB * 4);
  __bf16* x0h    = (__bf16*)wsa(NB * 2);
  __bf16* xch    = (__bf16*)wsa(NB * 2);
  float*  Zf     = (float*)wsa(4 * 20480 * 4);
  __bf16* Zh     = (__bf16*)wsa(4 * 20480 * 2);
  __bf16* Whp    = (__bf16*)wsa(128 * 128 * 2);
  __bf16* Mhp    = (__bf16*)wsa(64 * 128 * 2);
  float*  outacc = (float*)wsa((size_t)B_ * 64 * 4);
  (void)n_in; (void)out_size; (void)ws_size;

  sg_prep<<<1, 256, 0, stream>>>(hidden_adj, hidden_feat, fc_w, mlp_w, Zf, Zh, Whp, Mhp);
  sg_zero<<<(B_ * 64 + 255) / 256, 256, 0, stream>>>(outacc, (size_t)B_ * 64);
  sg_embed_fc<<<N_ / 16, 256, 0, stream>>>(x_idx, poi_emb, Whp, fc_b, xa, x0h);
  sg_zx_fuse<<<N_ / 16, 320, 0, stream>>>(Zh, x0h, x0h, batch, outacc, 0);

  float* xcur = xa;
  float* xnxt = xb;
  for (int i = 1; i < 4; ++i) {
    sg_zero4<<<(int)((NB / 4 + 255) / 256), 256, 0, stream>>>((float4*)xnxt, NB / 4);
    sg_mp<<<E_ / 8, 256, 0, stream>>>(edge_index, xcur, xnxt, E_);
    sg_cvt<<<(int)((NB / 4 + 255) / 256), 256, 0, stream>>>(xnxt, xch, NB / 4);
    sg_zx_fuse<<<N_ / 16, 320, 0, stream>>>(Zh, x0h, xch, batch, outacc, i);
    float* t = xcur; xcur = xnxt; xnxt = t;
  }
  sg_final<<<B_ / 16, 256, 0, stream>>>(outacc, Mhp, mlp_b, out);
}
